// PatchGraphAttention_43301860278554
// MI455X (gfx1250) — matrix operand order repaired and checked by reading, not compiled, not run on hardware
//
#include <hip/hip_runtime.h>
#include <hip/hip_bf16.h>
#include <stdint.h>

// ---------------------------------------------------------------------------
// CDNA5 (gfx1250) WMMA attention pipeline, wave32.
//   D_MODEL=1024, NHEAD=16, D_HEAD=64, B=4, M=2048
//   TAU=300, DELTA=300  => time bias = log(exp(-|i-j|) + 1e-12)
// Flash K/V tiles staged via Tensor Data Mover (tensor_load_to_lds) with
// TDM LDS padding producing the bank-conflict-free 72-element row stride.
// ---------------------------------------------------------------------------

typedef __attribute__((ext_vector_type(16))) __bf16 bfrag;   // 16x32 A / 32x16 B (8 VGPRs)
typedef __attribute__((ext_vector_type(8)))  float  ffrag;   // 16x16 f32 C/D (8 VGPRs)
typedef __attribute__((ext_vector_type(4)))  unsigned int v4u;
typedef __attribute__((ext_vector_type(8)))  int v8i;
typedef __attribute__((ext_vector_type(4)))  int v4i;

#define D_MODEL 1024
#define NHEAD   16
#define D_HEAD  64
#define BATCH   4
#define SEQ     2048
#define ROWS    (BATCH * SEQ)   // 8192

static __device__ inline ffrag fzero() {
  ffrag f;
#pragma unroll
  for (int i = 0; i < 8; ++i) f[i] = 0.0f;
  return f;
}

static __device__ inline ffrag wmma_bf16(bfrag a, bfrag b, ffrag c) {
  // D = A(16x32) * B(32x16) + C(16x16), fp32 accumulate
  return __builtin_amdgcn_wmma_f32_16x16x32_bf16(false, a, false, b, (short)0, c, false, false);
}

static __device__ inline unsigned short f2bf(float x) {
  uint32_t u = __float_as_uint(x);
  uint32_t r = (u + 0x7FFFu + ((u >> 16) & 1u)) >> 16;   // round-to-nearest-even
  return (unsigned short)r;
}

union FragU {
  bfrag v;
  uint4 q[2];
};

// A-fragment: lane l16 = l&15 selects row; two 8-elem groups
//   K = k0 + half*8 .. +7   and   K = k0 + 16 + half*8 .. +7
static __device__ inline bfrag ld_afrag(const unsigned short* rowPtrK0, int half) {
  FragU f;
  f.q[0] = *(const uint4*)(rowPtrK0 + half * 8);
  f.q[1] = *(const uint4*)(rowPtrK0 + 16 + half * 8);
  return f.v;
}

// B-fragment: lane holds column n = l16, K = k0 + half*8 .. +7 and k0 + 16 + half*8 .. +7
static __device__ inline bfrag ld_bfrag(const unsigned short* colRowPtrK0, int half) {
  FragU f;
  f.q[0] = *(const uint4*)(colRowPtrK0 + half * 8);
  f.q[1] = *(const uint4*)(colRowPtrK0 + 16 + half * 8);
  return f.v;
}

// ---------------------------------------------------------------------------
// Tensor Data Mover: 2D bf16 tile (rows x rowLen elems, global row stride
// rowStride elems) -> LDS at byte offset ldsOff, padding each 32-dword row
// chunk with 4 dwords => LDS row stride 72 bf16 elems (144 B).
// D# layout per CDNA5 ISA.
// ---------------------------------------------------------------------------
#if __has_builtin(__builtin_amdgcn_tensor_load_to_lds)
#define HAVE_TDM 1
static __device__ inline void tdm_load_tile(unsigned int ldsOff,
                                            const unsigned short* g,
                                            unsigned int rowLen,
                                            unsigned int rows,
                                            unsigned int rowStride) {
  const unsigned long long ga = (unsigned long long)(uintptr_t)g;
  v4u g0;
  g0[0] = 1u;                                              // count=1, user mode
  g0[1] = ldsOff;                                          // lds_addr
  g0[2] = (unsigned int)ga;                                // global_addr[31:0]
  g0[3] = (unsigned int)((ga >> 32) & 0x01FFFFFFull)       // global_addr[56:32]
          | (2u << 30);                                    // type=2 ("image")
  v8i g1;
  g1[0] = (int)((1u << 16)        // data_size = 2 bytes
                | (1u << 20)      // pad_enable
                | (4u << 22)      // pad_interval: 32 dwords
                | (3u << 25));    // pad_amount: 4 dwords
  g1[1] = (int)((rowLen & 0xFFFFu) << 16);                 // tensor_dim0 lo16
  g1[2] = (int)(((rowLen >> 16) & 0xFFFFu)                 // tensor_dim0 hi16
                | ((rows & 0xFFFFu) << 16));               // tensor_dim1 lo16
  g1[3] = (int)(((rows >> 16) & 0xFFFFu)                   // tensor_dim1 hi16
                | ((rowLen & 0xFFFFu) << 16));             // tile_dim0
  g1[4] = (int)(rows & 0xFFFFu);                           // tile_dim1 (tile_dim2=0)
  g1[5] = (int)rowStride;                                  // tensor_dim0_stride lo32
  g1[6] = 0;                                               // stride hi16, dim1_stride lo16
  g1[7] = 0;
  const v4i z4 = {0, 0, 0, 0};
#if defined(__clang_major__) && __clang_major__ >= 23
  const v8i z8 = {0, 0, 0, 0, 0, 0, 0, 0};
  __builtin_amdgcn_tensor_load_to_lds(g0, g1, z4, z4, z8, 0);
#else
  __builtin_amdgcn_tensor_load_to_lds(g0, g1, z4, z4, 0);
#endif
}
#else
#define HAVE_TDM 0
#endif

// ---------------------------------------------------------------------------
// fp32 -> bf16 conversion
// ---------------------------------------------------------------------------
__global__ void cvt_f32_bf16_kernel(const float* __restrict__ in,
                                    unsigned short* __restrict__ out, int n) {
  for (int i = blockIdx.x * blockDim.x + threadIdx.x; i < n;
       i += gridDim.x * blockDim.x)
    out[i] = f2bf(in[i]);
}

// ---------------------------------------------------------------------------
// Fused QKV projection:  X(8192x1024) @ W^T (+bias) -> scattered bf16 outputs
//   z==0 -> Q[b][h][m][d]   z==1 -> K[b][h][m][d]   z==2 -> Vt[b][h][d][m]
// Wave tile 64x32 (4x2 of 16x16), workgroup 8 waves -> 128x128 tile.
// ---------------------------------------------------------------------------
__global__ void __launch_bounds__(256)
qkv_gemm_kernel(const unsigned short* __restrict__ Hb,
                const unsigned short* __restrict__ Wqb,
                const unsigned short* __restrict__ Wkb,
                const unsigned short* __restrict__ Wvb,
                const float* __restrict__ bq, const float* __restrict__ bk,
                const float* __restrict__ bv,
                unsigned short* __restrict__ Qb,
                unsigned short* __restrict__ Kb,
                unsigned short* __restrict__ Vtb) {
  const int mode = blockIdx.z;
  const unsigned short* Wb = (mode == 0) ? Wqb : (mode == 1) ? Wkb : Wvb;
  const float* bias = (mode == 0) ? bq : (mode == 1) ? bk : bv;
  unsigned short* dst = (mode == 0) ? Qb : (mode == 1) ? Kb : Vtb;

  const int tid = threadIdx.x;
  const int lane = tid & 31, wave = tid >> 5;
  const int half = lane >> 4, l16 = lane & 15;
  const int baseM = blockIdx.x * 128 + (wave & 1) * 64;
  const int baseN = blockIdx.y * 128 + (wave >> 1) * 32;

  ffrag acc[4][2];
#pragma unroll
  for (int tr = 0; tr < 4; ++tr)
#pragma unroll
    for (int tn = 0; tn < 2; ++tn) acc[tr][tn] = fzero();

  const unsigned short* aRow[4];
#pragma unroll
  for (int tr = 0; tr < 4; ++tr)
    aRow[tr] = Hb + (size_t)(baseM + tr * 16 + l16) * D_MODEL;
  const unsigned short* bRow[2];
#pragma unroll
  for (int tn = 0; tn < 2; ++tn)
    bRow[tn] = Wb + (size_t)(baseN + tn * 16 + l16) * D_MODEL;

  for (int k0 = 0; k0 < D_MODEL; k0 += 32) {
    __builtin_prefetch(aRow[0] + k0 + 256, 0, 1);   // global_prefetch_b8
    bfrag af[4], bf[2];
#pragma unroll
    for (int tr = 0; tr < 4; ++tr) af[tr] = ld_afrag(aRow[tr] + k0, half);
#pragma unroll
    for (int tn = 0; tn < 2; ++tn) bf[tn] = ld_bfrag(bRow[tn] + k0, half);
#pragma unroll
    for (int tr = 0; tr < 4; ++tr)
#pragma unroll
      for (int tn = 0; tn < 2; ++tn)
        acc[tr][tn] = wmma_bf16(af[tr], bf[tn], acc[tr][tn]);
  }

#pragma unroll
  for (int tn = 0; tn < 2; ++tn) {
    const int col = baseN + tn * 16 + l16;
    const float bcol = bias[col];
    const int hh = col >> 6, dd = col & 63;
#pragma unroll
    for (int tr = 0; tr < 4; ++tr) {
#pragma unroll
      for (int r = 0; r < 8; ++r) {
        const int row = baseM + tr * 16 + half * 8 + r;
        const int bb = row >> 11, mm = row & 2047;
        const float val = acc[tr][tn][r] + bcol;
        if (mode < 2)
          dst[(((size_t)(bb * NHEAD + hh) * SEQ + mm) << 6) + dd] = f2bf(val);
        else
          dst[((size_t)(bb * NHEAD + hh) * D_HEAD + dd) * SEQ + mm] = f2bf(val);
      }
    }
  }
}

// ---------------------------------------------------------------------------
// Flash attention per (bh, q-tile).  Workgroup = 4 waves x 16 q rows = 64 rows.
// K/V streamed through LDS in 64-key tiles (TDM with pad -> 72-elem stride);
// online softmax in registers; P transposed through per-wave LDS staging.
// ---------------------------------------------------------------------------
#define KSTRIDE 72   // padded LDS row stride (bf16 elems)

__global__ void __launch_bounds__(128)
flash_attn_kernel(const unsigned short* __restrict__ Qb,
                  const unsigned short* __restrict__ Kb,
                  const unsigned short* __restrict__ Vtb,
                  const int* __restrict__ pmask,
                  unsigned short* __restrict__ AOb) {
  __shared__ unsigned short sK[64 * KSTRIDE];          // [key][d]
  __shared__ unsigned short sV[64 * KSTRIDE];          // [d][key]
  __shared__ unsigned short sP[4][16 * KSTRIDE];       // per-wave P staging

  const int tid = threadIdx.x;
  const int lane = tid & 31, wave = tid >> 5;
  const int half = lane >> 4, l16 = lane & 15;
  const int bh = blockIdx.y;
  const int b = bh >> 4, h = bh & 15;
  const int q0 = blockIdx.x * 64 + wave * 16;          // this wave's 16 q rows

  // Q fragments for the whole kernel: 16 rows x 64 dh = 2 A-fragments
  const unsigned short* qRow = Qb + ((size_t)bh * SEQ + q0 + l16) * D_HEAD;
  bfrag qf[2];
  qf[0] = ld_afrag(qRow + 0, half);
  qf[1] = ld_afrag(qRow + 32, half);

  ffrag oacc[4];
#pragma unroll
  for (int j = 0; j < 4; ++j) oacc[j] = fzero();
  float mrow[8], lrow[8];
#pragma unroll
  for (int r = 0; r < 8; ++r) { mrow[r] = -3.0e38f; lrow[r] = 0.0f; }

  const int* pmB = pmask + b * SEQ;
#if HAVE_TDM
  const unsigned int ldsK = (unsigned int)(uintptr_t)(void*)&sK[0];
  const unsigned int ldsV = (unsigned int)(uintptr_t)(void*)&sV[0];
#endif

  for (int kt = 0; kt < SEQ / 64; ++kt) {
    const int key0 = kt * 64;
    __syncthreads();           // previous tile fully consumed
#if HAVE_TDM
    if (wave == 0) {
      // K tile: 64 rows (keys) x 64 bf16, row stride 64 elems
      tdm_load_tile(ldsK, Kb + ((size_t)bh * SEQ + key0) * D_HEAD, 64, 64, 64);
      // Vt tile: 64 rows (d) x 64 bf16 (keys), row stride SEQ elems
      tdm_load_tile(ldsV, Vtb + (size_t)bh * D_HEAD * SEQ + key0, 64, 64, SEQ);
      __builtin_amdgcn_s_wait_tensorcnt(0);   // tiles resident in LDS
    }
#else
#pragma unroll
    for (int i = 0; i < 4; ++i) {
      const int idx = tid + i * 128;                 // 512 uint4 per tile
      const int row = idx >> 3, c = idx & 7;
      *(uint4*)&sK[row * KSTRIDE + c * 8] =
          *(const uint4*)(Kb + ((size_t)bh * SEQ + key0 + row) * D_HEAD + c * 8);
      *(uint4*)&sV[row * KSTRIDE + c * 8] =
          *(const uint4*)(Vtb + ((size_t)bh * D_HEAD + row) * SEQ + key0 + c * 8);
    }
#endif
    __syncthreads();

    // --- S = Q @ K^T for 4 key sub-tiles of 16 ---
    ffrag s[4];
#pragma unroll
    for (int j = 0; j < 4; ++j) s[j] = fzero();
#pragma unroll
    for (int ks = 0; ks < 2; ++ks) {
#pragma unroll
      for (int j = 0; j < 4; ++j) {
        bfrag bk = ld_bfrag(&sK[(j * 16 + l16) * KSTRIDE + ks * 32], half);
        s[j] = wmma_bf16(qf[ks], bk, s[j]);
      }
    }

    // --- scale + time bias + key mask ---
    int km[4];
#pragma unroll
    for (int j = 0; j < 4; ++j) km[j] = pmB[key0 + j * 16 + l16];
#pragma unroll
    for (int j = 0; j < 4; ++j) {
      const int ki = key0 + j * 16 + l16;
      const float madd = (km[j] > 0) ? 0.0f : -1.0e9f;
#pragma unroll
      for (int r = 0; r < 8; ++r) {
        const int qi = q0 + half * 8 + r;
        const float dist = fabsf((float)(qi - ki));
        const float bias = __logf(__expf(-dist) + 1e-12f);
        s[j][r] = s[j][r] * 0.125f + bias + madd;
      }
    }

    // --- online softmax: row reductions across 16-lane half groups ---
#pragma unroll
    for (int r = 0; r < 8; ++r) {
      float v = s[0][r];
#pragma unroll
      for (int j = 1; j < 4; ++j) v = fmaxf(v, s[j][r]);
      v = fmaxf(v, __shfl_xor(v, 1, 16));
      v = fmaxf(v, __shfl_xor(v, 2, 16));
      v = fmaxf(v, __shfl_xor(v, 4, 16));
      v = fmaxf(v, __shfl_xor(v, 8, 16));
      const float mnew = fmaxf(mrow[r], v);
      const float alpha = __expf(mrow[r] - mnew);
      float rs = 0.0f;
#pragma unroll
      for (int j = 0; j < 4; ++j) {
        const float p = __expf(s[j][r] - mnew);
        s[j][r] = p;
        rs += p;
      }
      rs += __shfl_xor(rs, 1, 16);
      rs += __shfl_xor(rs, 2, 16);
      rs += __shfl_xor(rs, 4, 16);
      rs += __shfl_xor(rs, 8, 16);
      lrow[r] = lrow[r] * alpha + rs;
      mrow[r] = mnew;
#pragma unroll
      for (int j = 0; j < 4; ++j) oacc[j][r] *= alpha;
    }

    // --- P (C layout) -> bf16 in per-wave LDS, reread as A fragments ---
#pragma unroll
    for (int j = 0; j < 4; ++j)
#pragma unroll
      for (int r = 0; r < 8; ++r)
        sP[wave][(half * 8 + r) * KSTRIDE + j * 16 + l16] = f2bf(s[j][r]);

#pragma unroll
    for (int kk = 0; kk < 2; ++kk) {
      bfrag pa = ld_afrag(&sP[wave][l16 * KSTRIDE + kk * 32], half);
#pragma unroll
      for (int j = 0; j < 4; ++j) {
        bfrag bv = ld_bfrag(&sV[(j * 16 + l16) * KSTRIDE + kk * 32], half);
        oacc[j] = wmma_bf16(pa, bv, oacc[j]);
      }
    }
  }

  // --- normalize and write attention output as bf16 [b][m][h*64+d] ---
#pragma unroll
  for (int r = 0; r < 8; ++r) {
    const float inv = 1.0f / (lrow[r] + 1e-20f);
    const int m = q0 + half * 8 + r;
#pragma unroll
    for (int j = 0; j < 4; ++j) {
      const int d = h * D_HEAD + j * 16 + l16;
      AOb[((size_t)(b * SEQ + m)) * D_MODEL + d] = f2bf(oacc[j][r] * inv);
    }
  }
}

// ---------------------------------------------------------------------------
// Output projection: AO(8192x1024) @ Wo^T + bo, fp32 out, * patch_mask row
// ---------------------------------------------------------------------------
__global__ void __launch_bounds__(256)
out_gemm_kernel(const unsigned short* __restrict__ AOb,
                const unsigned short* __restrict__ Wob,
                const float* __restrict__ bo, const int* __restrict__ pmask,
                float* __restrict__ out) {
  const int tid = threadIdx.x;
  const int lane = tid & 31, wave = tid >> 5;
  const int half = lane >> 4, l16 = lane & 15;
  const int baseM = blockIdx.x * 128 + (wave & 1) * 64;
  const int baseN = blockIdx.y * 128 + (wave >> 1) * 32;

  ffrag acc[4][2];
#pragma unroll
  for (int tr = 0; tr < 4; ++tr)
#pragma unroll
    for (int tn = 0; tn < 2; ++tn) acc[tr][tn] = fzero();

  const unsigned short* aRow[4];
#pragma unroll
  for (int tr = 0; tr < 4; ++tr)
    aRow[tr] = AOb + (size_t)(baseM + tr * 16 + l16) * D_MODEL;
  const unsigned short* bRow[2];
#pragma unroll
  for (int tn = 0; tn < 2; ++tn)
    bRow[tn] = Wob + (size_t)(baseN + tn * 16 + l16) * D_MODEL;

  for (int k0 = 0; k0 < D_MODEL; k0 += 32) {
    __builtin_prefetch(aRow[0] + k0 + 256, 0, 1);
    bfrag af[4], bf[2];
#pragma unroll
    for (int tr = 0; tr < 4; ++tr) af[tr] = ld_afrag(aRow[tr] + k0, half);
#pragma unroll
    for (int tn = 0; tn < 2; ++tn) bf[tn] = ld_bfrag(bRow[tn] + k0, half);
#pragma unroll
    for (int tr = 0; tr < 4; ++tr)
#pragma unroll
      for (int tn = 0; tn < 2; ++tn)
        acc[tr][tn] = wmma_bf16(af[tr], bf[tn], acc[tr][tn]);
  }

#pragma unroll
  for (int tn = 0; tn < 2; ++tn) {
    const int col = baseN + tn * 16 + l16;
    const float bcol = bo[col];
#pragma unroll
    for (int tr = 0; tr < 4; ++tr) {
#pragma unroll
      for (int r = 0; r < 8; ++r) {
        const int row = baseM + tr * 16 + half * 8 + r;
        const float msk = (float)pmask[row];    // row = b*SEQ + m
        out[(size_t)row * D_MODEL + col] = (acc[tr][tn][r] + bcol) * msk;
      }
    }
  }
}

// ---------------------------------------------------------------------------
// Host launcher
// ---------------------------------------------------------------------------
extern "C" void kernel_launch(void* const* d_in, const int* in_sizes, int n_in,
                              void* d_out, int out_size, void* d_ws,
                              size_t ws_size, hipStream_t stream) {
  const float* H = (const float*)d_in[0];
  const int* pmask = (const int*)d_in[1];
  const float* Wq = (const float*)d_in[2];
  const float* bq = (const float*)d_in[3];
  const float* Wk = (const float*)d_in[4];
  const float* bk = (const float*)d_in[5];
  const float* Wv = (const float*)d_in[6];
  const float* bv = (const float*)d_in[7];
  const float* Wo = (const float*)d_in[8];
  const float* bo = (const float*)d_in[9];
  float* out = (float*)d_out;

  char* ws = (char*)d_ws;
  const size_t actSz = (size_t)ROWS * D_MODEL * sizeof(unsigned short);   // 16 MB
  const size_t wSz = (size_t)D_MODEL * D_MODEL * sizeof(unsigned short);  // 2 MB
  unsigned short* Hb  = (unsigned short*)(ws);
  unsigned short* Qb  = (unsigned short*)(ws + actSz);
  unsigned short* Kb  = (unsigned short*)(ws + 2 * actSz);
  unsigned short* Vtb = (unsigned short*)(ws + 3 * actSz);
  unsigned short* AOb = (unsigned short*)(ws + 4 * actSz);
  unsigned short* Wqb = (unsigned short*)(ws + 5 * actSz);
  unsigned short* Wkb = (unsigned short*)(ws + 5 * actSz + wSz);
  unsigned short* Wvb = (unsigned short*)(ws + 5 * actSz + 2 * wSz);
  unsigned short* Wob = (unsigned short*)(ws + 5 * actSz + 3 * wSz);

  // 1) bf16 conversions
  cvt_f32_bf16_kernel<<<2048, 256, 0, stream>>>(H, Hb, ROWS * D_MODEL);
  cvt_f32_bf16_kernel<<<1024, 256, 0, stream>>>(Wq, Wqb, D_MODEL * D_MODEL);
  cvt_f32_bf16_kernel<<<1024, 256, 0, stream>>>(Wk, Wkb, D_MODEL * D_MODEL);
  cvt_f32_bf16_kernel<<<1024, 256, 0, stream>>>(Wv, Wvb, D_MODEL * D_MODEL);
  cvt_f32_bf16_kernel<<<1024, 256, 0, stream>>>(Wo, Wob, D_MODEL * D_MODEL);

  // 2) fused QKV projection (z: 0=Q, 1=K, 2=V-transposed)
  dim3 gQKV(ROWS / 128, D_MODEL / 128, 3);
  qkv_gemm_kernel<<<gQKV, 256, 0, stream>>>(Hb, Wqb, Wkb, Wvb, bq, bk, bv, Qb,
                                            Kb, Vtb);

  // 3) flash attention: (q-tiles, b*h)
  dim3 gAtt(SEQ / 64, BATCH * NHEAD, 1);
  flash_attn_kernel<<<gAtt, 128, 0, stream>>>(Qb, Kb, Vtb, pmask, AOb);

  // 4) output projection with bias + mask epilogue (fp32 out)
  dim3 gOut(ROWS / 128, D_MODEL / 128, 1);
  out_gemm_kernel<<<gOut, 256, 0, stream>>>(AOb, Wob, bo, pmask, out);
}
